// EfficientWindowAttention_24532853194963
// MI455X (gfx1250) — compile-verified, matrix operand order checked by reading, not run on hardware
//
#include <hip/hip_runtime.h>
#include <hip/hip_bf16.h>
#include <math.h>

typedef __attribute__((ext_vector_type(16))) __bf16 v16bf;
typedef __attribute__((ext_vector_type(8)))  __bf16 v8bf;
typedef __attribute__((ext_vector_type(8)))  float  v8f;
typedef __attribute__((ext_vector_type(4)))  unsigned v4u;
typedef __attribute__((ext_vector_type(8)))  int      v8i;
typedef __attribute__((ext_vector_type(4)))  int      v4i;

#define DIM  256
#define NH   8
#define HD   32
#define WS2  64
#define BW   2048   // B * nW = 8 * 256
#define LDA  272    // padded LDS stride (elements); 272*2B = 544B, 16B multiple

__device__ __forceinline__ unsigned short f2bf_bits(float f) {
  union { float f; unsigned u; } c; c.f = f;
  unsigned u = c.u;
  u += 0x7FFFu + ((u >> 16) & 1u);   // round-to-nearest-even
  return (unsigned short)(u >> 16);
}

__device__ __forceinline__ v16bf frag_combine(v8bf lo, v8bf hi) {
  v16bf f;
#pragma unroll
  for (int i = 0; i < 8; ++i) { f[i] = lo[i]; f[8 + i] = hi[i]; }
  return f;
}

// Row-pattern fragment (A tile, or B tile of a transposed operand): 16x32 bf16,
// row-major, stride lda (multiple of 8). Two 16-byte loads per lane.
__device__ __forceinline__ v16bf load_frag_rowK(const unsigned short* src, int lda, int lane) {
  const unsigned short* p0 = src + (lane & 15) * lda + ((lane >> 4) & 1) * 8;
  return frag_combine(*(const v8bf*)p0, *(const v8bf*)(p0 + 16));
}

// Pre-swizzled fragment: 512 bf16 stored in exact per-lane order; 32 contiguous B/lane.
__device__ __forceinline__ v16bf load_frag_sw(const unsigned short* tile, int lane) {
  const unsigned short* p0 = tile + lane * 16;
  return frag_combine(*(const v8bf*)p0, *(const v8bf*)(p0 + 8));
}

__device__ __forceinline__ v8f wmma_bf16(v16bf a, v16bf b, v8f c) {
  return __builtin_amdgcn_wmma_f32_16x16x32_bf16(false, a, false, b, (short)0, c, false, false);
}

// ---------------- Tensor Data Mover: build D# groups and issue tensor_load_to_lds.
// This toolchain (clang-23 / therock-10.0) exposes the 6-arg builtin:
//   (uint32x4 g0, int32x8 g1, int32x4 g2, int32x4 g3, int32x8 extra, i32 cpol)
__device__ __forceinline__ void tdm_load(unsigned ldsAddr, unsigned long long ga,
                                         unsigned dsz,                        // data_size code
                                         unsigned td0, unsigned td1, unsigned td2,
                                         unsigned long long s0, unsigned long long s1,
                                         unsigned tl0, unsigned tl1, unsigned tl2,
                                         unsigned padi, unsigned pada, unsigned pade) {
#if __has_builtin(__builtin_amdgcn_tensor_load_to_lds)
  v4u g0; v8i g1; v4i g2; v4i g3; v8i g4;
  g0[0] = 1u;                                            // count=1 (valid user D#)
  g0[1] = ldsAddr;                                       // lds_addr
  g0[2] = (unsigned)(ga & 0xFFFFFFFFull);                // global_addr[31:0]
  g0[3] = (unsigned)((ga >> 32) & 0x01FFFFFFull) | (2u << 30);  // [56:32] | type=2
  g1[0] = (int)((dsz << 16) | (pade << 20) | (padi << 22) | (pada << 25));
  g1[1] = (int)((td0 & 0xFFFFu) << 16);
  g1[2] = (int)((td0 >> 16) | ((td1 & 0xFFFFu) << 16));
  g1[3] = (int)((td1 >> 16) | (tl0 << 16));
  g1[4] = (int)((tl1 & 0xFFFFu) | (tl2 << 16));
  g1[5] = (int)(s0 & 0xFFFFFFFFull);
  g1[6] = (int)(((s0 >> 32) & 0xFFFFull) | ((s1 & 0xFFFFull) << 16));
  g1[7] = (int)((s1 >> 16) & 0xFFFFFFFFull);
  g2[0] = (int)td2; g2[1] = 0; g2[2] = 0; g2[3] = 0;
  g3[0] = 0; g3[1] = 0; g3[2] = 0; g3[3] = 0;
  g4[0] = 0; g4[1] = 0; g4[2] = 0; g4[3] = 0;
  g4[4] = 0; g4[5] = 0; g4[6] = 0; g4[7] = 0;
  __builtin_amdgcn_tensor_load_to_lds(g0, g1, g2, g3, g4, 0);
  __builtin_amdgcn_s_wait_tensorcnt(0);
#endif
}

// ---------------------------------------------------------------- kernel 0: weights -> swizzled bf16
// Store each 32x16 B-tile in per-lane fragment order so GEMM B loads are 32 contiguous B/lane.
__global__ __launch_bounds__(256) void wconv_kernel(const float* __restrict__ Wqkv,
                                                    const float* __restrict__ Wproj,
                                                    unsigned short* __restrict__ Wq_sw,
                                                    unsigned short* __restrict__ Wp_sw) {
  const int idx = blockIdx.x * 256 + threadIdx.x;
  const int e = idx & 15, lane = (idx >> 4) & 31, tile = idx >> 9;
  const int half = lane >> 4, n = lane & 15;
  const int p = e >> 1, j = e & 1;
  const int kk = ((p < 4) ? 0 : 16) + half * 8 + 2 * (p & 3) + j;
  const int ks = tile & 7, nt = tile >> 3;
  const int row = ks * 32 + kk;
  if (idx < 48 * 8 * 512) Wq_sw[idx] = f2bf_bits(Wqkv[row * 768 + nt * 16 + n]);
  if (idx < 16 * 8 * 512) Wp_sw[idx] = f2bf_bits(Wproj[row * 256 + nt * 16 + n]);
}

// ---------------------------------------------------------------- kernel 1: window partition + QKV GEMM
__global__ __launch_bounds__(256) void qkv_kernel(const float* __restrict__ x,
                                                  const unsigned short* __restrict__ Wq_sw,
                                                  const float* __restrict__ bqkv,
                                                  unsigned short* __restrict__ q,
                                                  unsigned short* __restrict__ k,
                                                  unsigned short* __restrict__ v) {
  extern __shared__ char smem[];
  float*          ldsF = (float*)smem;                       // 64 x 272 fp32  (69632 B)
  unsigned short* ldsB = (unsigned short*)(smem + 69632);    // 64 x 272 bf16  (34816 B)

  const int bw = blockIdx.x;
  const int b  = bw >> 8;
  const int w  = bw & 255;
  const int wh = w >> 4, ww = w & 15;
  const int tid = threadIdx.x;

#if __has_builtin(__builtin_amdgcn_tensor_load_to_lds)
  // TDM: 3D tile (C=256 contig) x (8 cols, stride 256) x (8 rows, stride 128*256),
  // pad +16 dwords per 256-dword row -> LDS stride 272 fp32.
  if (tid < 32) {
    const unsigned long long ga =
        (unsigned long long)(size_t)(x + ((size_t)b * 16384 + wh * 1024 + ww * 8) * DIM);
    tdm_load((unsigned)(size_t)(void*)ldsF, ga, /*dsz=*/2,
             /*td0=*/256, /*td1=*/128, /*td2=*/128,
             /*s0=*/256ull, /*s1=*/32768ull,
             /*tl0=*/256, /*tl1=*/8, /*tl2=*/8,
             /*padi=*/7, /*pada=*/15, /*pade=*/1);
  }
#else
  for (int t = 0; t < WS2; ++t) {
    const int rr = t >> 3, cc = t & 7;
    const int n = (wh * 8 + rr) * 128 + (ww * 8 + cc);
    ldsF[t * LDA + tid] = x[((size_t)b * 16384 + n) * DIM + tid];
  }
#endif
  __syncthreads();

  // fp32 -> bf16 pass (packed u32 stores)
  for (int i = 0; i < 32; ++i) {
    const int t  = i * 2 + (tid >> 7);
    const int c2 = (tid & 127) * 2;
    const float2 xv = *(const float2*)&ldsF[t * LDA + c2];
    *(unsigned*)&ldsB[t * LDA + c2] =
        (unsigned)f2bf_bits(xv.x) | ((unsigned)f2bf_bits(xv.y) << 16);
  }
  __syncthreads();

  const int wave = tid >> 5, lane = tid & 31;
  const int half = lane >> 4, nl = lane & 15;

  for (int g = 0; g < 6; ++g) {
    const int nt = wave + 8 * g;
    const int n0 = nt * 16;
    v8f acc[4] = {};
#pragma unroll
    for (int ks = 0; ks < 8; ++ks) {
      __builtin_prefetch(Wq_sw + ((size_t)(nt * 8 + ks + 1) << 9), 0, 1);
      const v16bf bf = load_frag_sw(Wq_sw + ((size_t)(nt * 8 + ks) << 9), lane);
#pragma unroll
      for (int mt = 0; mt < 4; ++mt) {
        const v16bf af = load_frag_rowK(ldsB + (mt * 16) * LDA + ks * 32, LDA, lane);
        acc[mt] = wmma_bf16(af, bf, acc[mt]);
      }
    }
    const int col = n0 + nl;
    const int which = col >> 8;
    const int h = (col >> 5) & 7;
    const int d = col & 31;
    const float bb = bqkv[col];
    if (which == 2) {
      // V: store in B-fragment swizzled order (4 tiles of 512 per head).
#pragma unroll
      for (int mt = 0; mt < 4; ++mt) {
#pragma unroll
        for (int r = 0; r < 8; ++r) {
          const int t = mt * 16 + r + 8 * half;
          const int kk = t & 31, ks2 = t >> 5, dt = d >> 4;
          const int vh_ = (kk >> 3) & 1;
          const int pp = ((kk >> 1) & 3) + (((kk >> 4) & 1) << 2);
          const int e = 2 * pp + (kk & 1);
          const size_t idx = ((((size_t)(bw * NH + h) * 2 + ks2) * 2 + dt) << 9) +
                             (((d & 15) + (vh_ << 4)) << 4) + e;
          v[idx] = f2bf_bits(acc[mt][r] + bb);
        }
      }
    } else {
      unsigned short* op = which ? k : q;      // rows layout (bw,h,t,d)
#pragma unroll
      for (int mt = 0; mt < 4; ++mt) {
#pragma unroll
        for (int r = 0; r < 8; ++r) {
          const int t = mt * 16 + r + 8 * half;
          op[(((size_t)bw * NH + h) * WS2 + t) * HD + d] = f2bf_bits(acc[mt][r] + bb);
        }
      }
    }
  }
}

// ---------------------------------------------------------------- kernel 2: fused attention per window
__global__ __launch_bounds__(256) void attn_kernel(const float* __restrict__ mask,
                                                   const unsigned short* __restrict__ q,
                                                   const unsigned short* __restrict__ k,
                                                   const unsigned short* __restrict__ v,
                                                   unsigned short* __restrict__ x2) {
  extern __shared__ unsigned short pls[];          // 8 heads * 64 * 80 bf16 = 81920 B
  const int bw  = blockIdx.x;
  const int w   = bw & 255;
  const int tid = threadIdx.x;
  const int h   = tid >> 5;
  const int lane = tid & 31;
  const int half = lane >> 4, nl = lane & 15;

  const size_t hb = ((size_t)bw * NH + h) * WS2 * HD;
  const unsigned short* qh = q + hb;
  const unsigned short* kh = k + hb;
  const unsigned short* vsw = v + (((size_t)(bw * NH + h)) << 11);  // 4 swizzled 512-elem tiles
  unsigned short* P = pls + h * WS2 * 80;

  const float scale = 0.17677669529663687f;        // 1/sqrt(32)
  const float slope = exp2f(-(float)(h + 1));      // alibi slope 0.5^(h+1)
  const float* mw = mask + (size_t)w * WS2 * WS2;

  // S = q @ k^T : K = hd = 32 -> single WMMA K-step; all operands 2x b128 per lane.
  v16bf aq[4];
#pragma unroll
  for (int mt = 0; mt < 4; ++mt) aq[mt] = load_frag_rowK(qh + mt * 16 * HD, HD, lane);
  v8f S[4][4] = {};
#pragma unroll
  for (int nt = 0; nt < 4; ++nt) {
    const v16bf bk = load_frag_rowK(kh + nt * 16 * HD, HD, lane);
#pragma unroll
    for (int mt = 0; mt < 4; ++mt) S[mt][nt] = wmma_bf16(aq[mt], bk, S[mt][nt]);
  }

  // scale + alibi + shift-mask, register softmax over keys, P -> LDS (bf16).
#pragma unroll
  for (int mt = 0; mt < 4; ++mt) {
#pragma unroll
    for (int r = 0; r < 8; ++r) {
      const int M = mt * 16 + r + 8 * half;
      float sv[4];
      float vmax = -3.0e38f;
#pragma unroll
      for (int nt = 0; nt < 4; ++nt) {
        const int N = nt * 16 + nl;
        const float val = S[mt][nt][r] * scale - slope * fabsf((float)(M - N)) + mw[M * 64 + N];
        sv[nt] = val;
        vmax = fmaxf(vmax, val);
      }
#pragma unroll
      for (int off = 1; off < 16; off <<= 1) vmax = fmaxf(vmax, __shfl_xor(vmax, off, 32));
      float sum = 0.f;
#pragma unroll
      for (int nt = 0; nt < 4; ++nt) { sv[nt] = __expf(sv[nt] - vmax); sum += sv[nt]; }
#pragma unroll
      for (int off = 1; off < 16; off <<= 1) sum += __shfl_xor(sum, off, 32);
      const float rs = 1.0f / sum;
#pragma unroll
      for (int nt = 0; nt < 4; ++nt) P[M * 80 + nt * 16 + nl] = f2bf_bits(sv[nt] * rs);
    }
  }
  __syncthreads();

  // O = P @ v : K = 64 keys (2 steps), N = 32 (2 tiles). V is pre-swizzled.
#pragma unroll
  for (int dt = 0; dt < 2; ++dt) {
    v8f O[4] = {};
#pragma unroll
    for (int ks = 0; ks < 2; ++ks) {
      const v16bf bv = load_frag_sw(vsw + ((size_t)(ks * 2 + dt) << 9), lane);
#pragma unroll
      for (int mt = 0; mt < 4; ++mt) {
        const v16bf ap = load_frag_rowK(P + (mt * 16) * 80 + ks * 32, 80, lane);
        O[mt] = wmma_bf16(ap, bv, O[mt]);
      }
    }
#pragma unroll
    for (int mt = 0; mt < 4; ++mt) {
#pragma unroll
      for (int r = 0; r < 8; ++r) {
        const int t = mt * 16 + r + 8 * half;
        x2[((size_t)bw * WS2 + t) * DIM + h * HD + dt * 16 + nl] = f2bf_bits(O[mt][r]);
      }
    }
  }
}

// ---------------------------------------------------------------- kernel 3: proj GEMM + window reverse
__global__ __launch_bounds__(256) void proj_kernel(const unsigned short* __restrict__ x2,
                                                   const unsigned short* __restrict__ Wp_sw,
                                                   const float* __restrict__ bproj,
                                                   float* __restrict__ out) {
  extern __shared__ unsigned short a2[];           // 64 x 272 bf16 (34816 B)
  const int bw = blockIdx.x;
  const int b  = bw >> 8;
  const int w  = bw & 255;
  const int wh = w >> 4, ww = w & 15;
  const int tid = threadIdx.x;

#if __has_builtin(__builtin_amdgcn_tensor_load_to_lds)
  // TDM: contiguous 2D tile 64 x 256 bf16, pad +8 dwords per 128-dword row (stride 272).
  if (tid < 32) {
    const unsigned long long ga = (unsigned long long)(size_t)(x2 + (size_t)bw * WS2 * DIM);
    tdm_load((unsigned)(size_t)(void*)a2, ga, /*dsz=*/1,
             /*td0=*/256, /*td1=*/(unsigned)(WS2 * BW), /*td2=*/1,
             /*s0=*/256ull, /*s1=*/0ull,
             /*tl0=*/256, /*tl1=*/64, /*tl2=*/0,
             /*padi=*/6, /*pada=*/7, /*pade=*/1);
  }
#else
  for (int i = 0; i < 16; ++i) {
    const int t = i * 4 + (tid >> 6);
    const int c = (tid & 63) * 4;
    *(uint2*)&a2[t * LDA + c] = *(const uint2*)&x2[((size_t)bw * WS2 + t) * DIM + c];
  }
#endif
  __syncthreads();

  const int wave = tid >> 5, lane = tid & 31;
  const int half = lane >> 4, nl = lane & 15;

  for (int g = 0; g < 2; ++g) {
    const int nt = wave + 8 * g;
    const int n0 = nt * 16;
    v8f acc[4] = {};
#pragma unroll
    for (int ks = 0; ks < 8; ++ks) {
      __builtin_prefetch(Wp_sw + ((size_t)(nt * 8 + ks + 1) << 9), 0, 1);
      const v16bf bf = load_frag_sw(Wp_sw + ((size_t)(nt * 8 + ks) << 9), lane);
#pragma unroll
      for (int mt = 0; mt < 4; ++mt) {
        const v16bf af = load_frag_rowK(a2 + (mt * 16) * LDA + ks * 32, LDA, lane);
        acc[mt] = wmma_bf16(af, bf, acc[mt]);
      }
    }
    const int col = n0 + nl;
    const float bb = bproj[col];
#pragma unroll
    for (int mt = 0; mt < 4; ++mt) {
#pragma unroll
      for (int r = 0; r < 8; ++r) {
        const int t = mt * 16 + r + 8 * half;
        const int rr = t >> 3, cc = t & 7;
        const int n = (wh * 8 + rr) * 128 + (ww * 8 + cc);
        out[((size_t)b * 16384 + n) * DIM + col] = acc[mt][r] + bb;
      }
    }
  }
}

extern "C" void kernel_launch(void* const* d_in, const int* in_sizes, int n_in,
                              void* d_out, int out_size, void* d_ws, size_t ws_size,
                              hipStream_t stream) {
  (void)in_sizes; (void)n_in; (void)out_size; (void)ws_size;
  const float* x     = (const float*)d_in[0];
  const float* mask  = (const float*)d_in[1];
  const float* Wqkv  = (const float*)d_in[2];
  const float* bqkv  = (const float*)d_in[3];
  const float* Wproj = (const float*)d_in[4];
  const float* bproj = (const float*)d_in[5];
  float* out = (float*)d_out;

  char* ws = (char*)d_ws;
  const size_t perQKV = (size_t)BW * NH * WS2 * HD * 2;   // 64 MiB each (bf16)
  unsigned short* q  = (unsigned short*)(ws);
  unsigned short* k  = (unsigned short*)(ws + perQKV);
  unsigned short* v  = (unsigned short*)(ws + 2 * perQKV);
  unsigned short* x2 = (unsigned short*)(ws + 3 * perQKV);
  unsigned short* Wq_sw = (unsigned short*)(ws + 3 * perQKV + (size_t)BW * WS2 * DIM * 2);
  unsigned short* Wp_sw = Wq_sw + 48 * 8 * 512;

  wconv_kernel<<<768, 256, 0, stream>>>(Wqkv, Wproj, Wq_sw, Wp_sw);
  qkv_kernel<<<BW, 256, 69632 + 34816, stream>>>(x, Wq_sw, bqkv, q, k, v);
  attn_kernel<<<BW, 256, NH * WS2 * 80 * 2, stream>>>(mask, q, k, v, x2);
  proj_kernel<<<BW, 256, WS2 * LDA * 2, stream>>>(x2, Wp_sw, bproj, out);
}
